// MNIST_SNN_59691455480586
// MI455X (gfx1250) — compile-verified, matrix operand order checked by reading, not run on hardware
//
#include <hip/hip_runtime.h>

typedef _Float16 half8 __attribute__((ext_vector_type(8)));
typedef _Float16 v16h  __attribute__((ext_vector_type(16)));
typedef float    v8f   __attribute__((ext_vector_type(8)));
typedef unsigned int u32x4 __attribute__((ext_vector_type(4)));
typedef int      i32x4 __attribute__((ext_vector_type(4)));
typedef int      i32x8 __attribute__((ext_vector_type(8)));

// ---- f16 weight scratch layout in d_ws (halves) ----
#define W1H_OFF 0                         // 256 x 800 (784 padded to 800 with zeros)
#define W2H_OFF (256 * 800)               // 128 x 256
#define W3H_OFF (256 * 800 + 128 * 256)   // 16 x 128 (rows 10..15 zero)

#define T_STEPS 100
#define BATCH   1024

// Tensor Data Mover path: probe-confirmed builtins on gfx1250 toolchains.
#if defined(__AMDGCN__) && __has_builtin(__builtin_amdgcn_tensor_load_to_lds) && \
    __has_builtin(__builtin_amdgcn_s_wait_tensorcnt)
#define USE_TDM 1
#endif

// ---------------- weight fp32 -> f16 conversion (runs once per launch) -------
__global__ void snn_cvt_weights(const float* __restrict__ W1,
                                const float* __restrict__ W2,
                                const float* __restrict__ W3,
                                _Float16* __restrict__ wsH) {
    int id = blockIdx.x * blockDim.x + threadIdx.x;
    const int n1 = 256 * 800, n2 = 128 * 256, n3 = 16 * 128;
    if (id < n1) {
        int n = id / 800, k = id % 800;
        wsH[W1H_OFF + id] = (k < 784) ? (_Float16)W1[n * 784 + k] : (_Float16)0.0f;
    } else if (id < n1 + n2) {
        int i = id - n1;
        wsH[W2H_OFF + i] = (_Float16)W2[i];
    } else if (id < n1 + n2 + n3) {
        int i = id - n1 - n2;
        int n = i / 128, k = i % 128;
        wsH[W3H_OFF + i] = (n < 10) ? (_Float16)W3[n * 128 + k] : (_Float16)0.0f;
    }
}

// Assemble a 16-half WMMA operand from two 16B chunks (K 0..7 / 16..23 per lane half)
__device__ __forceinline__ v16h ld_frag(const _Float16* p) {
    half8 lo = *(const half8*)p;
    half8 hi = *(const half8*)(p + 16);
    return __builtin_shufflevector(lo, hi, 0, 1, 2, 3, 4, 5, 6, 7,
                                           8, 9, 10, 11, 12, 13, 14, 15);
}

__device__ __forceinline__ v8f wmma32(v16h a, v16h b, v8f c) {
    return __builtin_amdgcn_wmma_f32_16x16x32_f16(false, a, false, b,
                                                  (short)0, c, false, false);
}

#ifdef USE_TDM
// Issue one TDM descriptor: copy a contiguous 16x784 f32 tile (50176 B) from
// global memory into LDS at byte offset ldsOff. 2D D# per ISA ch.8:
//   group0: count=1 | lds_addr | global_addr | type=2
//   group1: data_size=4B, tensor_dim0=784, tensor_dim1=16, tile 784x16, stride 784
__device__ __forceinline__ void tdm_load_tile(const float* gsrc, unsigned ldsOff) {
    unsigned long long ga = (unsigned long long)gsrc;
    u32x4 g0;
    g0.x = 1u;                                                   // count = 1
    g0.y = ldsOff;                                               // lds_addr (bytes)
    g0.z = (unsigned)ga;                                         // global_addr[31:0]
    g0.w = (unsigned)((ga >> 32) & 0x01FFFFFFull) | 0x80000000u; // addr[56:32] | type=2
    i32x8 g1;
    g1[0] = (2 << 16);          // workgroup_mask=0, data_size=2 (4 bytes)
    g1[1] = (int)(784u << 16);  // tensor_dim0[15:0] = 784
    g1[2] = (int)(16u << 16);   // tensor_dim0[31:16]=0, tensor_dim1[15:0]=16
    g1[3] = (int)(784u << 16);  // tensor_dim1[31:16]=0, tile_dim0=784
    g1[4] = 16;                 // tile_dim1=16, tile_dim2=0
    g1[5] = 784;                // tensor_dim0_stride[31:0] = 784
    g1[6] = 0;                  // stride hi, dim1_stride lo
    g1[7] = 0;
    i32x4 z4 = {0, 0, 0, 0};
#if __clang_major__ >= 23
    i32x8 z8 = {0, 0, 0, 0, 0, 0, 0, 0};
    __builtin_amdgcn_tensor_load_to_lds(g0, g1, z4, z4, z8, 0);
#else
    __builtin_amdgcn_tensor_load_to_lds(g0, g1, z4, z4, 0);
#endif
}
#endif

// ---------------- persistent SNN kernel: 64 WGs x 16 batch rows --------------
__launch_bounds__(256)
__global__ void snn_persistent(const float* __restrict__ spk_in,
                               const float* __restrict__ b1,
                               const float* __restrict__ b2,
                               const float* __restrict__ b3,
                               const _Float16* __restrict__ wsH,
                               float* __restrict__ out) {
    // LDS: padded strides chosen so row stride mod 64 banks == 4 dwords
    __shared__ __align__(16) _Float16 xs[16 * 800];    // x_t tile, f16, zero-padded K
    __shared__ __align__(16) _Float16 s1h[16 * 264];   // layer1 spikes (stride 264)
    __shared__ __align__(16) _Float16 s2h[16 * 136];   // layer2 spikes (stride 136)
    __shared__ __align__(16) _Float16 w2s[128 * 264];  // W2 f16 (stride 264)
    __shared__ __align__(16) _Float16 w3s[16 * 136];   // W3 f16 padded (stride 136)
#ifdef USE_TDM
    __shared__ __align__(16) float    xbuf[2][16 * 784]; // TDM double buffer (f32)
#endif

    const int tid   = threadIdx.x;
    const int wave  = tid >> 5;
    const int lane  = tid & 31;
    const int lr    = lane & 15;          // A-row / B-col / C-col within tile
    const int khalf = (lane >> 4) << 3;   // K sub-offset: 0 or 8
    const int r0    = blockIdx.x * 16;    // batch rows [r0, r0+16)

    // Stage W2 / W3 into LDS with padded strides (before first barrier)
    const _Float16* W1H = wsH + W1H_OFF;
    const _Float16* W2H = wsH + W2H_OFF;
    const _Float16* W3H = wsH + W3H_OFF;
    for (int i = tid; i < 128 * 256; i += 256) w2s[(i >> 8) * 264 + (i & 255)] = W2H[i];
    for (int i = tid; i < 16 * 128;  i += 256) w3s[(i >> 7) * 136 + (i & 127)] = W3H[i];
    // zero the K-pad columns (784..799) of xs once; they never change
    { int zr = tid >> 4, zc = 784 + (tid & 15); xs[zr * 800 + zc] = (_Float16)0.0f; }

#ifdef USE_TDM
    // prologue: start DMA of the t=0 tile into buffer 0 (wave 0 issues)
    if (wave == 0) {
        tdm_load_tile(spk_in + (size_t)r0 * 784,
                      (unsigned)(size_t)(const void*)&xbuf[0][0]);
    }
#endif

    // Persistent membrane state in accumulator registers
    v8f m1a = {}, m1b = {};   // layer1: two 16x16 N-tiles per wave (N = 256)
    v8f m2  = {};             // layer2: one 16x16 N-tile per wave (N = 128)
    v8f m3  = {};             // layer3: wave 0 only (N padded 10 -> 16)

    // Per-lane bias broadcasts (constant over time)
    const float bv1a = b1[wave * 32 + lr];
    const float bv1b = b1[wave * 32 + 16 + lr];
    const float bv2  = b2[wave * 16 + lr];
    const float bv3  = (lr < 10) ? b3[lr] : 0.0f;

    const int row_m = khalf;  // C-matrix row base: M = r + 8*(lane>>4)

    for (int t = 0; t < T_STEPS; ++t) {
        const float* xt = spk_in + (size_t)t * (BATCH * 784) + (size_t)r0 * 784;
#ifdef USE_TDM
        if (wave == 0) __builtin_amdgcn_s_wait_tensorcnt((short)0); // buf[t&1] landed
        __syncthreads();      // all waves see buf[t&1]; prior-step xs reads done
        if (wave == 0 && (t + 1) < T_STEPS) {
            // start DMA of next step's tile into the other buffer (hidden behind
            // this whole step's WMMA work)
            tdm_load_tile(xt + (size_t)(BATCH * 784),
                          (unsigned)(size_t)(const void*)&xbuf[(t + 1) & 1][0]);
        }
        // convert LDS f32 tile -> f16 xs, 8 elements per chunk (b128 traffic)
        {
            const float* src = &xbuf[t & 1][0];
            for (int c = tid; c < 16 * 98; c += 256) {
                int row = c / 98, col = (c - row * 98) * 8;
                const float* sp = src + row * 784 + col;
                half8 h;
                #pragma unroll
                for (int i = 0; i < 8; ++i) h[i] = (_Float16)sp[i];
                *(half8*)&xs[row * 800 + col] = h;
            }
        }
#else
        __syncthreads();      // prior-step xs reads done
        // fallback staging: global f32 -> f16 LDS, vectorized, plus prefetch
        for (int c = tid; c < 16 * 98; c += 256) {
            int row = c / 98, col = (c - row * 98) * 8;
            const float* sp = xt + row * 784 + col;
            half8 h;
            #pragma unroll
            for (int i = 0; i < 8; ++i) h[i] = (_Float16)sp[i];
            *(half8*)&xs[row * 800 + col] = h;
        }
        if (t + 1 < T_STEPS) {
            const float* xn = xt + (size_t)(BATCH * 784);
            for (int p = tid; p < 392; p += 256)   // 392 * 128B = 16*784*4B
                __builtin_prefetch(xn + p * 32, 0, 3);
        }
#endif
        __syncthreads();  // barrier: xs (and, at t==0, w2s/w3s) ready

        // ---- layer 1: cur1 = x @ W1^T, two N-tiles per wave, K = 800 ----
        {
            v8f acc0, acc1;
            #pragma unroll
            for (int r = 0; r < 8; ++r) {
                float mp0 = m1a[r], mp1 = m1b[r];
                acc0[r] = 0.9f * mp0 - ((mp0 > 1.0f) ? 1.0f : 0.0f) + bv1a;
                acc1[r] = 0.9f * mp1 - ((mp1 > 1.0f) ? 1.0f : 0.0f) + bv1b;
            }
            const _Float16* ap  = &xs[lr * 800 + khalf];
            const _Float16* bp0 = W1H + (size_t)(wave * 32 + lr) * 800 + khalf;
            const _Float16* bp1 = W1H + (size_t)(wave * 32 + 16 + lr) * 800 + khalf;
            #pragma unroll 5
            for (int k0 = 0; k0 < 800; k0 += 32) {
                v16h a  = ld_frag(ap + k0);
                v16h w0 = ld_frag(bp0 + k0);
                v16h w1 = ld_frag(bp1 + k0);
                acc0 = wmma32(a, w0, acc0);
                acc1 = wmma32(a, w1, acc1);
            }
            m1a = acc0;
            m1b = acc1;
            #pragma unroll
            for (int r = 0; r < 8; ++r) {
                int m = r + row_m;
                s1h[m * 264 + wave * 32 + lr]      =
                    (_Float16)(((acc0[r] - 1.0f) > 0.0f) ? 1.0f : 0.0f);
                s1h[m * 264 + wave * 32 + 16 + lr] =
                    (_Float16)(((acc1[r] - 1.0f) > 0.0f) ? 1.0f : 0.0f);
            }
        }
        __syncthreads();  // barrier: s1 ready

        // ---- layer 2: cur2 = s1 @ W2^T, one N-tile per wave, K = 256 ----
        {
            v8f acc;
            #pragma unroll
            for (int r = 0; r < 8; ++r) {
                float mp = m2[r];
                acc[r] = 0.9f * mp - ((mp > 1.0f) ? 1.0f : 0.0f) + bv2;
            }
            const _Float16* ap = &s1h[lr * 264 + khalf];
            const _Float16* bp = &w2s[(wave * 16 + lr) * 264 + khalf];
            #pragma unroll
            for (int k0 = 0; k0 < 256; k0 += 32) {
                v16h a = ld_frag(ap + k0);
                v16h w = ld_frag(bp + k0);
                acc = wmma32(a, w, acc);
            }
            m2 = acc;
            #pragma unroll
            for (int r = 0; r < 8; ++r) {
                int m = r + row_m;
                s2h[m * 136 + wave * 16 + lr] =
                    (_Float16)(((acc[r] - 1.0f) > 0.0f) ? 1.0f : 0.0f);
            }
        }
        __syncthreads();  // barrier: s2 ready

        // ---- layer 3: cur3 = s2 @ W3^T, wave 0 only, K = 128, N padded to 16 ----
        if (wave == 0) {
            v8f acc;
            #pragma unroll
            for (int r = 0; r < 8; ++r) {
                float mp = m3[r];
                acc[r] = 0.9f * mp - ((mp > 1.0f) ? 1.0f : 0.0f) + bv3;
            }
            const _Float16* ap = &s2h[lr * 136 + khalf];
            const _Float16* bp = &w3s[lr * 136 + khalf];
            #pragma unroll
            for (int k0 = 0; k0 < 128; k0 += 32) {
                v16h a = ld_frag(ap + k0);
                v16h w = ld_frag(bp + k0);
                acc = wmma32(a, w, acc);
            }
            m3 = acc;
            if (lr < 10) {
                float* spk_rec = out + (size_t)t * (BATCH * 10);
                float* mem_rec = out + (size_t)T_STEPS * BATCH * 10 + (size_t)t * (BATCH * 10);
                #pragma unroll
                for (int r = 0; r < 8; ++r) {
                    int m   = r + row_m;
                    int row = r0 + m;
                    spk_rec[row * 10 + lr] = ((acc[r] - 1.0f) > 0.0f) ? 1.0f : 0.0f;
                    mem_rec[row * 10 + lr] = acc[r];
                }
            }
        }
        // No trailing barrier needed: next-step writes never race prior reads
        // (xs reads of step t all precede the s1 barrier of step t; the TDM for
        // t+1 targets the opposite xbuf parity).
    }
}

extern "C" void kernel_launch(void* const* d_in, const int* in_sizes, int n_in,
                              void* d_out, int out_size, void* d_ws, size_t ws_size,
                              hipStream_t stream) {
    const float* spk_in = (const float*)d_in[0];
    const float* W1     = (const float*)d_in[1];
    const float* b1     = (const float*)d_in[2];
    const float* W2     = (const float*)d_in[3];
    const float* b2     = (const float*)d_in[4];
    const float* W3     = (const float*)d_in[5];
    const float* b3     = (const float*)d_in[6];
    float*          out = (float*)d_out;
    _Float16*       wsH = (_Float16*)d_ws;

    // 1) convert weights to f16 padded layouts in scratch (~480 KB)
    const int total = 256 * 800 + 128 * 256 + 16 * 128;
    snn_cvt_weights<<<(total + 255) / 256, 256, 0, stream>>>(W1, W2, W3, wsH);

    // 2) persistent SNN: 64 workgroups x 16 batch rows, full T-loop on chip
    snn_persistent<<<64, 256, 0, stream>>>(spk_in, b1, b2, b3, wsH, out);
}
